// LstmCloseModel_26860725469411
// MI455X (gfx1250) — matrix operand order repaired and checked by reading, not compiled, not run on hardware
//
#include <hip/hip_runtime.h>
#include <hip/hip_bf16.h>
#include <stddef.h>
#include <stdint.h>

// ---------------- problem constants ----------------
constexpr int T_STEPS = 365;
constexpr int NGRID   = 3000;
constexpr int NXF     = 20;      // x features
constexpr int HDIM    = 256;     // hidden
constexpr int KDIM    = 512;     // concat [x0|h]
constexpr int NGATE   = 1024;    // 4*H
constexpr int ROWS    = 32;      // grid rows per workgroup (two WMMA M tiles)
constexpr int KT      = KDIM / 32;   // 16 WMMA K-steps
constexpr int NTILES  = NGATE / 16;  // 64 N tiles

typedef __attribute__((ext_vector_type(16))) _Float16 v16h;
typedef __attribute__((ext_vector_type(8)))  _Float16 v8h;
typedef __attribute__((ext_vector_type(8)))  float    v8f;

// LDS layout (bytes)
constexpr int OFF_ABUF = 0;                       // [32][512] f16   = 32768
constexpr int OFF_WIN  = 32768;                   // [256][21] f32   = 21504
constexpr int OFF_XCAT = 32768 + 21504;           // [32][21]  f32   =  2688
constexpr int OFF_YPRV = OFF_XCAT + 2688;         // [32]      f32   =   128
constexpr int OFF_PART = OFF_YPRV + 128;          // [16][32]  f32   =  2048
constexpr int LDS_BYTES = OFF_PART + 2048;        // 59136

// ---------------------------------------------------------------------------
// Pack B = [w_ih | w_hh]^T (512 x 1024) into f16 WMMA B-fragment order.
// Fragment layout (ISA 7.12.2, 16-bit B 32x16): half i of lane L holds
//   B[kt*32 + 8*(L>>4) + (i<8 ? i : 8+i)][nt*16 + (L&15)]
// Linear index: ((kt*64 + nt)*512 + lane*16 + i)  -> per-(kt,nt) tile is a
// contiguous 1KB block; each lane's 16 halves are one contiguous 32B load.
// ---------------------------------------------------------------------------
__global__ void pack_weights_kernel(const float* __restrict__ w_ih,
                                    const float* __restrict__ w_hh,
                                    _Float16* __restrict__ pB) {
    int idx  = blockIdx.x * 256 + threadIdx.x;      // < 16*64*512 = 524288
    int i    = idx & 15;
    int lane = (idx >> 4) & 31;
    int nt   = (idx >> 9) & 63;
    int kt   = idx >> 15;
    int k    = kt * 32 + ((lane >> 4) << 3) + ((i < 8) ? i : 8 + i);
    int n    = nt * 16 + (lane & 15);
    float v  = (k < HDIM) ? w_ih[n * HDIM + k] : w_hh[n * HDIM + (k - HDIM)];
    pB[idx]  = (_Float16)v;
}

// fast sigmoid / tanh: single v_exp_f32 + v_rcp_f32 each
__device__ __forceinline__ float sigf(float v) {
    return __builtin_amdgcn_rcpf(1.0f + __builtin_amdgcn_exp2f(-1.44269504f * v));
}
__device__ __forceinline__ float tanh_fast(float v) {
    // tanh(x) = 2/(1+exp(-2x)) - 1
    return fmaf(2.0f, __builtin_amdgcn_rcpf(1.0f + __builtin_amdgcn_exp2f(-2.88539008f * v)), -1.0f);
}

// ---------------------------------------------------------------------------
// Persistent LSTM scan: one workgroup owns 32 grid rows (2 M-tiles) for all
// 365 steps. 16 waves; wave wv owns N-tiles { gate*16 + wv : gate in 0..3 },
// i.e. columns j in [wv*16, wv*16+16) of EACH gate block. The four gate
// values for a given (row,j) land in the same lane/slot of acc[0..3] (per
// M-tile), so the cell update is fully register-resident; c never leaves
// VGPRs. Each B fragment load feeds TWO WMMAs (both M-tiles).
// ---------------------------------------------------------------------------
__global__ __launch_bounds__(512, 1)
void lstm_scan_kernel(const float* __restrict__ xg,   const float* __restrict__ yg,
                      const float* __restrict__ w_in, const float* __restrict__ b_in,
                      const float* __restrict__ b_ih, const float* __restrict__ b_hh,
                      const float* __restrict__ w_out,const float* __restrict__ b_out,
                      const _Float16* __restrict__ pB, float* __restrict__ out) {
    extern __shared__ char smem[];
    _Float16* Abuf     = (_Float16*)(smem + OFF_ABUF);   // [32][512] f16
    float*    w_inL    = (float*)(smem + OFF_WIN);       // [256][21]
    float*    xcatL    = (float*)(smem + OFF_XCAT);      // [32][21]
    float*    yprevL   = (float*)(smem + OFF_YPRV);      // [32]
    float*    partialL = (float*)(smem + OFF_PART);      // [16][32]

    const int tid  = threadIdx.x;        // 0..511
    const int lane = tid & 31;
    const int wv   = tid >> 5;           // 0..15
    const int base = blockIdx.x * ROWS;

    // --- stage constants ---
    for (int e = tid; e < HDIM * (NXF + 1); e += 512) w_inL[e] = w_in[e];

    const int xcol  = tid & 255;         // x0 column this thread owns
    const int xhalf = tid >> 8;          // which 16-row half of x0 it computes
    const float binv  = b_in[xcol];
    const float boutv = b_out[0];

    float bgr[4];                        // combined gate bias at this lane's column
    #pragma unroll
    for (int g4 = 0; g4 < 4; ++g4) {
        int col = g4 * 256 + wv * 16 + (lane & 15);
        bgr[g4] = b_ih[col] + b_hh[col];
    }
    const float woutv = w_out[wv * 16 + (lane & 15)];

    // --- state init ---
    v8f creg[2];
    #pragma unroll
    for (int mt = 0; mt < 2; ++mt)
        #pragma unroll
        for (int r = 0; r < 8; ++r) creg[mt][r] = 0.0f;

    for (int e = tid; e < ROWS * HDIM; e += 512) {            // h part of A := 0
        int r = e >> 8, j = e & 255;
        Abuf[r * KDIM + HDIM + j] = (_Float16)0.0f;
    }
    if (tid < ROWS) yprevL[tid] = 0.0f;
    __syncthreads();

    const int m      = lane & 15;                 // A row this lane supplies
    const int khalf  = (lane >> 4) << 3;          // 0 or 8 (A layout half-wave split)
    const int rowoff = (lane < 16) ? 0 : 8;       // C/D row offset

    for (int t = 0; t < T_STEPS; ++t) {
        // Blind an integer offset each iteration (uniform SGPR) so LICM cannot
        // hoist the per-step B loads out of the t-loop (which would spill
        // 2KB/lane of fragments to scratch). Adding the opaque offset to the
        // original pointer keeps the global address space (global_load, not
        // flat_load) and scalar-base addressing.
        int blind = 0;
        asm volatile("" : "+s"(blind));
        const _Float16* pBt = pB + blind;

        // ---- step 1: gather x_t and fillObs(y) into xcat ----
        for (int e = tid; e < ROWS * NXF; e += 512) {
            int r = e / NXF, k = e % NXF;
            int g = base + r;
            xcatL[r * (NXF + 1) + k] =
                (g < NGRID) ? xg[(size_t)t * NGRID * NXF + (size_t)g * NXF + k] : 0.0f;
        }
        if (tid < ROWS) {
            int g = base + tid;
            float yt = yprevL[tid];
            if (g < NGRID) {
                float yo = yg[(size_t)t * NGRID + g];
                if (!__builtin_isnan(yo)) yt = yo;
            }
            xcatL[tid * (NXF + 1) + NXF] = yt;
        }
        __syncthreads();

        // ---- step 2: x0 = relu(xcat @ w_in^T + b_in) ----
        // thread owns column xcol for 16 rows (xhalf selects which 16)
        #pragma unroll 4
        for (int rr = 0; rr < 16; ++rr) {
            int r = xhalf * 16 + rr;
            float a = binv;
            #pragma unroll
            for (int k = 0; k <= NXF; ++k)
                a = fmaf(xcatL[r * (NXF + 1) + k], w_inL[xcol * (NXF + 1) + k], a);
            Abuf[r * KDIM + xcol] = (_Float16)fmaxf(a, 0.0f);
        }
        __syncthreads();

        // ---- step 3: gates = A(32x512) x B(512x1024), WMMA f16->f32 ----
        // acc[g4*2 + mt]; each B fragment feeds both M-tiles.
        v8f acc[8];
        #pragma unroll
        for (int q = 0; q < 8; ++q)
            #pragma unroll
            for (int r = 0; r < 8; ++r) acc[q][r] = 0.0f;

        for (int kt = 0; kt < KT; ++kt) {
            int k0 = kt * 32 + khalf;
            v16h a0, a1;
            {
                v8h lo = *(const v8h*)(Abuf + m * KDIM + k0);
                v8h hi = *(const v8h*)(Abuf + m * KDIM + k0 + 16);
                #pragma unroll
                for (int i2 = 0; i2 < 8; ++i2) { a0[i2] = lo[i2]; a0[8 + i2] = hi[i2]; }
            }
            {
                v8h lo = *(const v8h*)(Abuf + (16 + m) * KDIM + k0);
                v8h hi = *(const v8h*)(Abuf + (16 + m) * KDIM + k0 + 16);
                #pragma unroll
                for (int i2 = 0; i2 < 8; ++i2) { a1[i2] = lo[i2]; a1[8 + i2] = hi[i2]; }
            }

            const _Float16* pkt = pBt + ((size_t)kt << 15) + (lane << 4);
            #pragma unroll
            for (int g4 = 0; g4 < 4; ++g4) {
                const _Float16* pb = pkt + ((size_t)(g4 * 16 + wv) << 9);
                v16h b = *(const v16h*)pb;
                acc[g4 * 2 + 0] = __builtin_amdgcn_wmma_f32_16x16x32_f16(
                    false, a0, false, b, (short)0, acc[g4 * 2 + 0], false, false);
                acc[g4 * 2 + 1] = __builtin_amdgcn_wmma_f32_16x16x32_f16(
                    false, a1, false, b, (short)0, acc[g4 * 2 + 1], false, false);
            }
        }

        // ---- step 4: LSTM cell, fully in registers ----
        v8f hreg[2];
        #pragma unroll
        for (int mt = 0; mt < 2; ++mt) {
            #pragma unroll
            for (int r = 0; r < 8; ++r) {
                float ig = sigf     (acc[0 * 2 + mt][r] + bgr[0]);
                float fg = sigf     (acc[1 * 2 + mt][r] + bgr[1]);
                float gg = tanh_fast(acc[2 * 2 + mt][r] + bgr[2]);
                float og = sigf     (acc[3 * 2 + mt][r] + bgr[3]);
                float c  = fg * creg[mt][r] + ig * gg;
                creg[mt][r] = c;
                hreg[mt][r] = og * tanh_fast(c);
            }
        }
        __syncthreads();   // all waves done reading Abuf before h overwrite

        // ---- step 5: write h (f16) back to A; y-partials via shuffles ----
        const int j = wv * 16 + (lane & 15);
        v8f part[2];
        #pragma unroll
        for (int mt = 0; mt < 2; ++mt) {
            #pragma unroll
            for (int r = 0; r < 8; ++r) {
                float h = hreg[mt][r];
                Abuf[(mt * 16 + r + rowoff) * KDIM + HDIM + j] = (_Float16)h;
                part[mt][r] = h * woutv;
            }
        }
        #pragma unroll
        for (int mt = 0; mt < 2; ++mt) {
            #pragma unroll
            for (int r = 0; r < 8; ++r) {
                float p = part[mt][r];
                p += __shfl_xor(p, 1, 32);
                p += __shfl_xor(p, 2, 32);
                p += __shfl_xor(p, 4, 32);
                p += __shfl_xor(p, 8, 32);
                part[mt][r] = p;
            }
        }
        if ((lane & 15) == 0) {
            #pragma unroll
            for (int mt = 0; mt < 2; ++mt)
                #pragma unroll
                for (int r = 0; r < 8; ++r)
                    partialL[wv * 32 + mt * 16 + rowoff + r] = part[mt][r];
        }
        __syncthreads();

        // ---- step 6: finalize y (NY=1), store, keep for next fillObs ----
        if (tid < ROWS) {
            float s = boutv;
            #pragma unroll
            for (int w2 = 0; w2 < 16; ++w2) s += partialL[w2 * 32 + tid];
            yprevL[tid] = s;
            int g = base + tid;
            if (g < NGRID) out[(size_t)t * NGRID + g] = s;
        }
        __syncthreads();
    }
}

extern "C" void kernel_launch(void* const* d_in, const int* in_sizes, int n_in,
                              void* d_out, int out_size, void* d_ws, size_t ws_size,
                              hipStream_t stream) {
    (void)in_sizes; (void)n_in; (void)out_size; (void)ws_size;
    const float* x     = (const float*)d_in[0];
    const float* y     = (const float*)d_in[1];
    const float* w_in  = (const float*)d_in[2];
    const float* b_in  = (const float*)d_in[3];
    const float* w_ih  = (const float*)d_in[4];
    const float* b_ih  = (const float*)d_in[5];
    const float* w_hh  = (const float*)d_in[6];
    const float* b_hh  = (const float*)d_in[7];
    const float* w_out = (const float*)d_in[8];
    const float* b_out = (const float*)d_in[9];
    float* out = (float*)d_out;
    _Float16* pB = (_Float16*)d_ws;              // 16*64*512 halves = 1 MB scratch

    // Repack [w_ih|w_hh]^T -> f16 WMMA B fragments (deterministic, idempotent).
    pack_weights_kernel<<<(KT * NTILES * 512) / 256, 256, 0, stream>>>(w_ih, w_hh, pB);

    const int nblk = (NGRID + ROWS - 1) / ROWS;  // 94
    lstm_scan_kernel<<<nblk, 512, LDS_BYTES, stream>>>(
        x, y, w_in, b_in, b_ih, b_hh, w_out, b_out, pB, out);
}
